// SurvPath_14620068675603
// MI455X (gfx1250) — matrix operand order repaired and checked by reading, not compiled, not run on hardware
//
#include <hip/hip_runtime.h>
#include <hip/hip_bf16.h>

typedef __attribute__((ext_vector_type(16))) __bf16 v16bf;
typedef __attribute__((ext_vector_type(8)))  float  v8f;
typedef int v4i __attribute__((__vector_size__(16)));
typedef __attribute__((address_space(1))) v4i* as1_v4i;
typedef __attribute__((address_space(3))) v4i* as3_v4i;

#define NPATCH 60000
#define TILE   128
#define LDW    264            // bf16 per LDS row: 256 + 8 pad (keeps 16B align, shifts banks)
#define NTASK  24             // 8 heads * 3 queries
#define NBLK   ((NPATCH + TILE - 1) / TILE)   // 469

// Native bf16 convert: let the backend select gfx1250's packed cvt instead of
// a 4-op manual RNE sequence (this sits on the WMMA critical path).
__device__ __forceinline__ unsigned short f2bfu(float f) {
    __bf16 h = (__bf16)f;
    return __builtin_bit_cast(unsigned short, h);
}
__device__ __forceinline__ float bf2f(unsigned short h) {
    unsigned u = ((unsigned)h) << 16;
    return __builtin_bit_cast(float, u);
}

// ---------------------------------------------------------------------------
// K0: wp [1024,256] fp32 -> wpt [256][1024] bf16 (transposed so B-matrix
// columns are contiguous for WMMA B-fragment loads)
// ---------------------------------------------------------------------------
__global__ __launch_bounds__(256) void k0_prep(const float* __restrict__ wp,
                                               unsigned short* __restrict__ wpt) {
    int t = blockIdx.x * 256 + threadIdx.x;    // over 1024*256
    if (t < 1024 * 256) {
        int k = t >> 8, n = t & 255;
        wpt[n * 1024 + k] = f2bfu(wp[k * 256 + n]);
    }
}

// ---------------------------------------------------------------------------
// K1: SNN pathways -> bag[6,256]; gene attention -> top-3; selected, q;
// QK[h,q,e] = (1/sqrt(32)) * sum_dh wk[e, h*32+dh] * q[q, h*32+dh];
// zero the pooled accumulator.
// ---------------------------------------------------------------------------
__global__ __launch_bounds__(256) void k1_genes(
    const float* xo0, const float* xo1, const float* xo2,
    const float* xo3, const float* xo4, const float* xo5,
    const float* w10, const float* w11, const float* w12,
    const float* w13, const float* w14, const float* w15,
    const float* __restrict__ sb1, const float* __restrict__ sw2,
    const float* __restrict__ sb2,
    const float* __restrict__ wa1, const float* __restrict__ ba1,
    const float* __restrict__ wa2, const float* __restrict__ ba2,
    const float* __restrict__ wq,  const float* __restrict__ bq,
    const float* __restrict__ wk,
    float* __restrict__ sel_out, float* __restrict__ qk_out,
    float* __restrict__ pool) {
    __shared__ float xb[600];
    __shared__ float h1[256];
    __shared__ float bag[6 * 256];
    __shared__ float qL[3 * 256];
    __shared__ float red[64];
    __shared__ float scores[6];
    __shared__ int   idx[3];
    const int t = threadIdx.x;
    const int   sizes[6] = {100, 200, 300, 400, 500, 600};
    const float* xs[6]  = {xo0, xo1, xo2, xo3, xo4, xo5};
    const float* w1s[6] = {w10, w11, w12, w13, w14, w15};

    for (int i = 0; i < 6; ++i) {
        int s = sizes[i];
        for (int k = t; k < s; k += 256) xb[k] = xs[i][k];
        __syncthreads();
        float a = sb1[i * 256 + t];
        for (int k = 0; k < s; ++k) a += xb[k] * w1s[i][k * 256 + t];
        h1[t] = (a > 0.f) ? a : (__expf(a) - 1.f);      // ELU
        __syncthreads();
        float b = sb2[i * 256 + t];
        const float* w2 = sw2 + (size_t)i * 256 * 256;
        for (int k = 0; k < 256; ++k) b += h1[k] * w2[k * 256 + t];
        bag[i * 256 + t] = (b > 0.f) ? b : (__expf(b) - 1.f);
        __syncthreads();
    }
    // gene attention scores (softmax is monotone -> top-k of raw scores)
    for (int i = 0; i < 6; ++i) {
        if (t < 64) {
            float a = ba1[t];
            for (int d = 0; d < 256; ++d) a += bag[i * 256 + d] * wa1[d * 64 + t];
            a = (a > 0.f) ? a : 0.f;                    // ReLU
            red[t] = a * wa2[t];
        }
        __syncthreads();
        if (t == 0) {
            float s = ba2[0];
            for (int j = 0; j < 64; ++j) s += red[j];
            scores[i] = s;
        }
        __syncthreads();
    }
    if (t == 0) {
        bool used[6] = {false, false, false, false, false, false};
        for (int kq = 0; kq < 3; ++kq) {
            int best = 0; float bv = -1e30f;
            for (int i = 0; i < 6; ++i)
                if (!used[i] && scores[i] > bv) { bv = scores[i]; best = i; }
            used[best] = true; idx[kq] = best;
        }
    }
    __syncthreads();
    for (int kq = 0; kq < 3; ++kq) sel_out[kq * 256 + t] = bag[idx[kq] * 256 + t];
    for (int kq = 0; kq < 3; ++kq) {
        float a = bq[t];
        const float* srow = bag + idx[kq] * 256;
        for (int e = 0; e < 256; ++e) a += srow[e] * wq[e * 256 + t];
        qL[kq * 256 + t] = a;
    }
    __syncthreads();
    const float scale = 0.17677669529663687f;           // 1/sqrt(32)
    {
        int e = t;
        for (int h = 0; h < 8; ++h)
            for (int qi = 0; qi < 3; ++qi) {
                float a = 0.f;
                for (int dh = 0; dh < 32; ++dh)
                    a += wk[e * 256 + h * 32 + dh] * qL[qi * 256 + h * 32 + dh];
                qk_out[(h * 3 + qi) * 256 + e] = a * scale;
            }
    }
    for (int i = t; i < 3 * 256; i += 256) pool[i] = 0.f;
}

// ---------------------------------------------------------------------------
// K2: per 128-row tile: wsi = relu(x @ wp + bp) via bf16 WMMA; pooled partial
// sums; flash-style block-local softmax stats + weighted context per (h,q).
// ---------------------------------------------------------------------------
__global__ __launch_bounds__(256) void k2_wsi_attn(
    const float* __restrict__ x,                // [N][1024]
    const unsigned short* __restrict__ wpt,     // [256][1024] bf16 (transposed)
    const float* __restrict__ bp,               // [256]
    const float* __restrict__ qk,               // [24][256], pre-scaled
    float* __restrict__ pool,                   // [3][256]
    float* __restrict__ stats,                  // [NBLK][24][2] (m, sumexp)
    float* __restrict__ ctxp,                   // [NBLK][24][256]
    int N) {
    extern __shared__ unsigned char smem[];
    unsigned short* wsi = (unsigned short*)smem;            // TILE*LDW bf16
    unsigned short* wch = wsi + TILE * LDW;                 // 256*32 bf16 (wp chunk)
    float* qks  = (float*)(wch + 256 * 32);                 // 24*256
    float* lrow = qks + NTASK * 256;                        // 24*TILE logits -> probs
    float* msb  = lrow + NTASK * TILE;                      // 24*2

    const int tid  = threadIdx.x;
    const int wave = tid >> 5;
    const int lane = tid & 31;
    const int r0   = blockIdx.x * TILE;
    const int nrows = (N - r0 < TILE) ? (N - r0) : TILE;

    for (int i = tid; i < NTASK * 256; i += 256) qks[i] = qk[i];

    // ----- wsi GEMM: rows wave*16..+16, all 256 cols, K = 1024 -----
    v8f acc[16];
#pragma unroll
    for (int ct = 0; ct < 16; ++ct) acc[ct] = (v8f){0.f,0.f,0.f,0.f,0.f,0.f,0.f,0.f};

    const int mrow = r0 + wave * 16 + (lane & 15);
    const int arow = (mrow < N) ? mrow : (N - 1);
    const float* xrow = x + (size_t)arow * 1024;
    const int kbA = (lane >> 4) * 16;       // A-frag K base per lane half
    const int nB  = lane & 15;              // B-frag column within tile
    const int kbB = (lane >> 4) * 16;       // B-frag K base per lane half

    for (int kc = 0; kc < 1024; kc += 32) {
        __syncthreads();                    // protect wch from previous iteration
        // stage wp chunk [32 K][256 N] -> LDS (thread tid owns column n=tid)
#if __has_builtin(__builtin_amdgcn_global_load_async_to_lds_b128)
        {
            as1_v4i gsrc = (as1_v4i)(wpt + (size_t)tid * 1024 + kc);
            as3_v4i ldst = (as3_v4i)(wch + tid * 32);
            __builtin_amdgcn_global_load_async_to_lds_b128(gsrc,     ldst,     0, 0);
            __builtin_amdgcn_global_load_async_to_lds_b128(gsrc + 1, ldst + 1, 0, 0);
            __builtin_amdgcn_global_load_async_to_lds_b128(gsrc + 2, ldst + 2, 0, 0);
            __builtin_amdgcn_global_load_async_to_lds_b128(gsrc + 3, ldst + 3, 0, 0);
#if __has_builtin(__builtin_amdgcn_s_wait_asynccnt)
            __builtin_amdgcn_s_wait_asynccnt(0);
#else
            asm volatile("s_wait_asynccnt 0x0" ::: "memory");
#endif
        }
#else
        {
            const uint4* src = (const uint4*)(wpt + (size_t)tid * 1024 + kc);
            uint4* dst = (uint4*)(wch + tid * 32);
            dst[0] = src[0]; dst[1] = src[1]; dst[2] = src[2]; dst[3] = src[3];
        }
#endif
        __syncthreads();
        // A fragment: 16x32 bf16, lane = row (lane&15), 16 contiguous K at kbA
        union { v16bf v; unsigned short u[16]; } A;
        const float* p0 = xrow + kc + kbA;
#pragma unroll
        for (int e = 0; e < 8; ++e) {
            A.u[e]     = f2bfu(p0[e]);
            A.u[8 + e] = f2bfu(p0[8 + e]);
        }
        if (kc + 32 < 1024) __builtin_prefetch(xrow + kc + 32, 0, 1);  // global_prefetch_b8
#pragma unroll
        for (int ct = 0; ct < 16; ++ct) {
            union { v16bf v; uint4 q[2]; } B;
            const uint4* bsrc = (const uint4*)(wch + (ct * 16 + nB) * 32 + kbB);
            B.q[0] = bsrc[0]; B.q[1] = bsrc[1];
            acc[ct] = __builtin_amdgcn_wmma_f32_16x16x32_bf16(
                false, A.v, false, B.v, (short)0, acc[ct], false, false);
        }
    }
    __syncthreads();
    // bias + relu, spill to LDS as bf16 (C layout: VGPR r -> row r + 8*(lane>=16))
    {
        const int rbase = wave * 16 + (lane >> 4) * 8;
#pragma unroll
        for (int ct = 0; ct < 16; ++ct) {
            const int col = ct * 16 + (lane & 15);
            const float b = bp[col];
#pragma unroll
            for (int r = 0; r < 8; ++r) {
                float v = acc[ct][r] + b;
                v = (v > 0.f) ? v : 0.f;
                wsi[(rbase + r) * LDW + col] = f2bfu(v);
            }
        }
    }
    __syncthreads();

    // ----- pooled (adaptive_avg_pool) partial sums; bins of 20000 rows -----
    {
        const int c = tid;
        float s = 0.f;
        int curb = r0 / 20000;
        for (int r = 0; r < nrows; ++r) {
            int b = (r0 + r) / 20000;
            if (b != curb) { atomicAdd(&pool[curb * 256 + c], s); s = 0.f; curb = b; }
            s += bf2f(wsi[r * LDW + c]);
        }
        atomicAdd(&pool[curb * 256 + c], s);
    }

    // ----- logits: l[task][row] = wsi[row] . qk[task] (scale folded in) -----
    if (tid < TILE) {
        const int r = tid;
        float lacc[NTASK];
#pragma unroll
        for (int tk = 0; tk < NTASK; ++tk) lacc[tk] = 0.f;
        if (r < nrows) {
            for (int e = 0; e < 256; ++e) {
                float w = bf2f(wsi[r * LDW + e]);
#pragma unroll
                for (int tk = 0; tk < NTASK; ++tk) lacc[tk] += w * qks[tk * 256 + e];
            }
            for (int tk = 0; tk < NTASK; ++tk) lrow[tk * TILE + r] = lacc[tk];
        } else {
            for (int tk = 0; tk < NTASK; ++tk) lrow[tk * TILE + r] = -1e30f;
        }
    }
    __syncthreads();
    // ----- block-local softmax stats; overwrite logits with probabilities ---
    if (tid < NTASK) {
        float m = -1e30f;
        for (int r = 0; r < nrows; ++r) m = fmaxf(m, lrow[tid * TILE + r]);
        float s = 0.f;
        for (int r = 0; r < TILE; ++r) {
            float p = (r < nrows) ? __expf(lrow[tid * TILE + r] - m) : 0.f;
            lrow[tid * TILE + r] = p;
            s += p;
        }
        msb[tid * 2] = m; msb[tid * 2 + 1] = s;
        stats[((size_t)blockIdx.x * NTASK + tid) * 2 + 0] = m;
        stats[((size_t)blockIdx.x * NTASK + tid) * 2 + 1] = s;
    }
    __syncthreads();
    // ----- weighted context partials: ctx[task][c] = sum_r p[r]*wsi[r][c] ---
    {
        const int c = tid;
        for (int tk = 0; tk < NTASK; ++tk) {
            float a = 0.f;
            for (int r = 0; r < nrows; ++r)
                a += lrow[tk * TILE + r] * bf2f(wsi[r * LDW + c]);
            ctxp[((size_t)blockIdx.x * NTASK + tk) * 256 + c] = a;
        }
    }
}

// ---------------------------------------------------------------------------
// K3: cross-block softmax merge (flash-attention merge) -> ctx_final[24][256]
// ---------------------------------------------------------------------------
__global__ __launch_bounds__(256) void k3_merge(const float* __restrict__ stats,
                                                const float* __restrict__ ctxp,
                                                float* __restrict__ ctxf, int NB) {
    __shared__ float red[256];
    __shared__ float Msh;
    const int T = blockIdx.x, t = threadIdx.x;
    float m = -1e30f;
    for (int b = t; b < NB; b += 256) m = fmaxf(m, stats[((size_t)b * NTASK + T) * 2]);
    red[t] = m; __syncthreads();
    for (int s = 128; s > 0; s >>= 1) {
        if (t < s) red[t] = fmaxf(red[t], red[t + s]);
        __syncthreads();
    }
    if (t == 0) Msh = red[0];
    __syncthreads();
    const float M = Msh;
    float acc = 0.f, S = 0.f;
    for (int b = 0; b < NB; ++b) {
        float w = __expf(stats[((size_t)b * NTASK + T) * 2] - M);
        S   += w * stats[((size_t)b * NTASK + T) * 2 + 1];
        acc += w * ctxp[((size_t)b * NTASK + T) * 256 + t];
    }
    ctxf[T * 256 + t] = acc / S;
}

// ---------------------------------------------------------------------------
// K4: head — V/O projections from contexts, pooled, LN1, fusion MLP, LN2, cls
// ---------------------------------------------------------------------------
__global__ __launch_bounds__(256) void k4_head(
    const float* __restrict__ ctxf, const float* __restrict__ wv,
    const float* __restrict__ bv,   const float* __restrict__ wo,
    const float* __restrict__ bo,   const float* __restrict__ pool,
    const float* __restrict__ sel,
    const float* __restrict__ ln1g, const float* __restrict__ ln1b,
    const float* __restrict__ wf1,  const float* __restrict__ bf1,
    const float* __restrict__ wf2,  const float* __restrict__ bf2,
    const float* __restrict__ ln2g, const float* __restrict__ ln2b,
    const float* __restrict__ wc,   const float* __restrict__ bc,
    float* __restrict__ out) {
    __shared__ float acat[3 * 256];
    __shared__ float y[3 * 256];
    __shared__ float att[3 * 256];
    __shared__ float am[256], gene[256], f1[256], fz[256];
    __shared__ float red[256];
    __shared__ float stat[2];
    const int t = threadIdx.x;
    const int h = t >> 5;                       // head owning output dim t
    // attn_cat[qi][t] = ctx[h(t)][qi] . wv[:,t] + bv[t]
    for (int qi = 0; qi < 3; ++qi) {
        float a = bv[t];
        const float* cx = ctxf + (h * 3 + qi) * 256;
        for (int e = 0; e < 256; ++e) a += cx[e] * wv[e * 256 + t];
        acat[qi * 256 + t] = a;
    }
    __syncthreads();
    for (int qi = 0; qi < 3; ++qi) {
        float a = bo[t];
        for (int e = 0; e < 256; ++e) a += acat[qi * 256 + e] * wo[e * 256 + t];
        y[qi * 256 + t] = a + pool[qi * 256 + t] * (1.f / 20000.f);
    }
    __syncthreads();
    for (int qi = 0; qi < 3; ++qi) {            // LayerNorm1 per query
        float x = y[qi * 256 + t];
        red[t] = x; __syncthreads();
        for (int s = 128; s > 0; s >>= 1) { if (t < s) red[t] += red[t + s]; __syncthreads(); }
        if (t == 0) stat[0] = red[0] * (1.f / 256.f);
        __syncthreads();
        float d = x - stat[0];
        red[t] = d * d; __syncthreads();
        for (int s = 128; s > 0; s >>= 1) { if (t < s) red[t] += red[t + s]; __syncthreads(); }
        if (t == 0) stat[1] = red[0] * (1.f / 256.f);
        __syncthreads();
        att[qi * 256 + t] = d * rsqrtf(stat[1] + 1e-5f) * ln1g[t] + ln1b[t];
        __syncthreads();
    }
    am[t]   = (att[t] + att[256 + t] + att[512 + t]) * (1.f / 3.f);
    gene[t] = (sel[t] + sel[256 + t] + sel[512 + t]) * (1.f / 3.f);
    __syncthreads();
    {
        float a = bf1[t];
        for (int e = 0; e < 256; ++e) a += am[e]   * wf1[e * 256 + t];
        for (int e = 0; e < 256; ++e) a += gene[e] * wf1[(256 + e) * 256 + t];
        f1[t] = (a > 0.f) ? a : 0.f;
    }
    __syncthreads();
    {
        float a = bf2[t];
        for (int e = 0; e < 256; ++e) a += f1[e] * wf2[e * 256 + t];
        a = (a > 0.f) ? a : 0.f;
        fz[t] = a + am[t];
    }
    __syncthreads();
    {   // LayerNorm2
        float x = fz[t];
        red[t] = x; __syncthreads();
        for (int s = 128; s > 0; s >>= 1) { if (t < s) red[t] += red[t + s]; __syncthreads(); }
        if (t == 0) stat[0] = red[0] * (1.f / 256.f);
        __syncthreads();
        float d = x - stat[0];
        red[t] = d * d; __syncthreads();
        for (int s = 128; s > 0; s >>= 1) { if (t < s) red[t] += red[t + s]; __syncthreads(); }
        if (t == 0) stat[1] = red[0] * (1.f / 256.f);
        __syncthreads();
        fz[t] = d * rsqrtf(stat[1] + 1e-5f) * ln2g[t] + ln2b[t];
    }
    __syncthreads();
    if (t < 4) {
        float a = bc[t];
        for (int d = 0; d < 256; ++d) a += fz[d] * wc[d * 4 + t];
        out[t] = a;
    }
}

// ---------------------------------------------------------------------------
extern "C" void kernel_launch(void* const* d_in, const int* in_sizes, int n_in,
                              void* d_out, int out_size, void* d_ws, size_t ws_size,
                              hipStream_t stream) {
    const float* x_path = (const float*)d_in[0];
    const float* xo[6]; for (int i = 0; i < 6; ++i) xo[i] = (const float*)d_in[1 + i];
    const float* w1[6]; for (int i = 0; i < 6; ++i) w1[i] = (const float*)d_in[7 + i];
    const float* sb1 = (const float*)d_in[13];
    const float* sw2 = (const float*)d_in[14];
    const float* sb2 = (const float*)d_in[15];
    const float* wp  = (const float*)d_in[16];
    const float* bp  = (const float*)d_in[17];
    const float* wa1 = (const float*)d_in[18];
    const float* ba1 = (const float*)d_in[19];
    const float* wa2 = (const float*)d_in[20];
    const float* ba2 = (const float*)d_in[21];
    const float* wq  = (const float*)d_in[22];
    const float* wk  = (const float*)d_in[23];
    const float* wv  = (const float*)d_in[24];
    const float* wo  = (const float*)d_in[25];
    const float* bq  = (const float*)d_in[26];
    /* bk (d_in[27]) cancels inside softmax — intentionally unused */
    const float* bv  = (const float*)d_in[28];
    const float* bo  = (const float*)d_in[29];
    const float* ln1g = (const float*)d_in[30];
    const float* ln1b = (const float*)d_in[31];
    const float* wf1 = (const float*)d_in[32];
    const float* bf1 = (const float*)d_in[33];
    const float* wf2 = (const float*)d_in[34];
    const float* bf2 = (const float*)d_in[35];
    const float* ln2g = (const float*)d_in[36];
    const float* ln2b = (const float*)d_in[37];
    const float* wc  = (const float*)d_in[38];
    const float* bc  = (const float*)d_in[39];

    char* ws = (char*)d_ws;
    size_t off = 0;
    unsigned short* wpt = (unsigned short*)(ws + off); off += (size_t)256 * 1024 * 2; // 512 KB
    float* qkbuf = (float*)(ws + off); off += (size_t)NTASK * 256 * 4;
    float* selbuf = (float*)(ws + off); off += (size_t)3 * 256 * 4;
    float* poolbuf = (float*)(ws + off); off += (size_t)3 * 256 * 4;
    float* statsbuf = (float*)(ws + off); off += (size_t)NBLK * NTASK * 2 * 4;
    float* ctxpbuf = (float*)(ws + off); off += (size_t)NBLK * NTASK * 256 * 4;
    float* ctxfbuf = (float*)(ws + off); off += (size_t)NTASK * 256 * 4;
    (void)ws_size; (void)in_sizes; (void)n_in; (void)out_size;

    k0_prep<<<1024, 256, 0, stream>>>(wp, wpt);

    k1_genes<<<1, 256, 0, stream>>>(
        xo[0], xo[1], xo[2], xo[3], xo[4], xo[5],
        w1[0], w1[1], w1[2], w1[3], w1[4], w1[5],
        sb1, sw2, sb2, wa1, ba1, wa2, ba2, wq, bq, wk,
        selbuf, qkbuf, poolbuf);

    const size_t smem2 = (size_t)TILE * LDW * 2      // wsi bf16
                       + (size_t)256 * 32 * 2        // wp chunk bf16
                       + (size_t)NTASK * 256 * 4     // qk
                       + (size_t)NTASK * TILE * 4    // logits/probs
                       + (size_t)NTASK * 2 * 4;      // m,s
    k2_wsi_attn<<<NBLK, 256, smem2, stream>>>(
        x_path, wpt, bp, qkbuf, poolbuf, statsbuf, ctxpbuf, NPATCH);

    k3_merge<<<NTASK, 256, 0, stream>>>(statsbuf, ctxpbuf, ctxfbuf, NBLK);

    k4_head<<<1, 256, 0, stream>>>(
        ctxfbuf, wv, bv, wo, bo, poolbuf, selbuf,
        ln1g, ln1b, wf1, bf1, wf2, bf2, ln2g, ln2b, wc, bc,
        (float*)d_out);
}
